// UMT5Attention_16827681866051
// MI455X (gfx1250) — matrix operand order repaired and checked by reading, not compiled, not run on hardware
//
#include <hip/hip_runtime.h>
#include <hip/hip_bf16.h>
#include <cstdint>

// ---------------------------------------------------------------------------
// UMT5 encoder self-attention for MI455X (gfx1250), bf16 WMMA pipeline.
//   B=2, S=2048, D=1024, H=16, DK=64, NUM_BUCKETS=32, MAX_DISTANCE=128
// Outputs (concatenated in d_out): attn_output [B,S,D] f32, attn [B,H,S,S] f32
// ---------------------------------------------------------------------------

typedef __attribute__((ext_vector_type(16))) __bf16 v16bf;
typedef __attribute__((ext_vector_type(8)))  __bf16 v8bfx;
typedef __attribute__((ext_vector_type(8)))  float  v8f;

#define B_  2
#define S_  2048
#define D_  1024
#define H_  16
#define DK_ 64

static __device__ __forceinline__ v8f wmma_bf16(v16bf a, v16bf b, v8f c) {
  return __builtin_amdgcn_wmma_f32_16x16x32_bf16(false, a, false, b, (short)0, c,
                                                 false, false);
}

static __device__ __forceinline__ float wave_max(float v) {
#pragma unroll
  for (int o = 16; o > 0; o >>= 1) v = fmaxf(v, __shfl_xor(v, o, 32));
  return v;
}
static __device__ __forceinline__ float wave_sum(float v) {
#pragma unroll
  for (int o = 16; o > 0; o >>= 1) v += __shfl_xor(v, o, 32);
  return v;
}

// ---------------- fp32 -> bf16 conversion (row-major keep) ----------------
__global__ void cvt_bf16_kernel(const float* __restrict__ in,
                                __bf16* __restrict__ out, int n) {
  int i = blockIdx.x * blockDim.x + threadIdx.x;
  if (i < n) out[i] = (__bf16)in[i];
}

// ---------------- fp32 [K][N] -> bf16 transposed [N][K] -------------------
__global__ void cvt_tr_bf16_kernel(const float* __restrict__ in,
                                   __bf16* __restrict__ out, int Kd, int Nd) {
  int i = blockIdx.x * blockDim.x + threadIdx.x;
  if (i >= Kd * Nd) return;
  int k = i / Nd, n = i - k * Nd;          // coalesced reads along n
  out[(size_t)n * Kd + k] = (__bf16)in[i];
}

// ---------------- relative-position bias table [H][4096] ------------------
// biasTab[h][d + 2047] = rel_bias[bucket(d)][h],  d = k - q in [-2047, 2047]
__global__ void bias_table_kernel(const float* __restrict__ rel_bias,
                                  float* __restrict__ biasTab) {
  int idx = blockIdx.x * blockDim.x + threadIdx.x;
  if (idx >= H_ * 4095) return;
  int h = idx / 4095;
  int d = (idx % 4095) - 2047;             // relative_position = mem - ctx
  int rb = (d > 0) ? 16 : 0;               // num_buckets//2 = 16
  int ad = d < 0 ? -d : d;
  int bucket;
  if (ad < 8) {                            // max_exact = 8
    bucket = ad;
  } else {
    float lr = __logf((float)ad * 0.125f) * (1.0f / __logf(16.0f));
    int lg = 8 + (int)(lr * 8.0f);
    bucket = lg < 15 ? lg : 15;
  }
  bucket += rb;
  biasTab[h * 4096 + (d + 2047)] = rel_bias[bucket * H_ + h];
}

// ---------------- LDS-free bf16 GEMM: C[M,N] = A[M,K] * Wt[N,K]^T ---------
// Block = 128 threads (4 waves); block tile 128x64; wave tile 32x64
// (2 row-groups x 4 col-tiles = 8 wmma accumulators). A and Wt fragments
// are contiguous per-lane global loads (weights stay hot in 192MB L2).
__global__ __launch_bounds__(128) void gemm_bf16_kernel(
    const __bf16* __restrict__ A, const __bf16* __restrict__ Wt,
    __bf16* __restrict__ Cb, float* __restrict__ Cf,
    int M, int N, int K, int writeF32) {
  const int tid  = threadIdx.x;
  const int lane = tid & 31;
  const int wvid = tid >> 5;
  const int m    = lane & 15;
  const int half = lane >> 4;
  const int kb   = half * 8;     // A-frag K sub-block  (ISA 16x32 bf16 layout)
  const int ko   = half * 16;    // contiguous-order K offset; unused (B frag uses kb, table order)
  const int mbase = blockIdx.x * 128;
  const int nbase = blockIdx.y * 64;
  const int rowA  = mbase + wvid * 32 + m;

  v8f acc[2][4];
#pragma unroll
  for (int rg = 0; rg < 2; ++rg)
#pragma unroll
    for (int nt = 0; nt < 4; ++nt)
#pragma unroll
      for (int j = 0; j < 8; ++j) acc[rg][nt][j] = 0.0f;

  for (int kk = 0; kk < K; kk += 32) {
    v16bf af[2];
#pragma unroll
    for (int rg = 0; rg < 2; ++rg) {
      const __bf16* ar = A + (size_t)(rowA + rg * 16) * K + kk;
      v8bfx lo = *(const v8bfx*)(ar + kb);
      v8bfx hi = *(const v8bfx*)(ar + 16 + kb);
#pragma unroll
      for (int j = 0; j < 8; ++j) { af[rg][j] = lo[j]; af[rg][8 + j] = hi[j]; }
    }
    if (kk + 32 < K)
      __builtin_prefetch(A + (size_t)rowA * K + kk + 32, 0, 1);
#pragma unroll
    for (int nt = 0; nt < 4; ++nt) {
      const __bf16* wr = Wt + (size_t)(nbase + nt * 16 + m) * K + kk;
      v8bfx blo = *(const v8bfx*)(wr + kb);
      v8bfx bhi = *(const v8bfx*)(wr + 16 + kb);
      v16bf bf;
#pragma unroll
      for (int j = 0; j < 8; ++j) { bf[j] = blo[j]; bf[8 + j] = bhi[j]; }
      acc[0][nt] = wmma_bf16(af[0], bf, acc[0][nt]);
      acc[1][nt] = wmma_bf16(af[1], bf, acc[1][nt]);
    }
  }

#pragma unroll
  for (int rg = 0; rg < 2; ++rg)
#pragma unroll
    for (int nt = 0; nt < 4; ++nt)
#pragma unroll
      for (int j = 0; j < 8; ++j) {
        int row = mbase + wvid * 32 + rg * 16 + j + 8 * half;
        int col = nbase + nt * 16 + m;
        if (writeF32) Cf[(size_t)row * N + col] = acc[rg][nt][j];
        else          Cb[(size_t)row * N + col] = (__bf16)acc[rg][nt][j];
      }
}

// ---------------- fused attention: scores+bias, softmax, P*V --------------
// One block per (b, h, 16-query tile). 128 threads = 4 waves.
#define SSTR 2052   // f32 row stride: 2052 mod 64 = 4 -> conflict-free gathers
#define VLD  72     // sV row stride (bf16)
#define SS_BYTES (16 * SSTR * 4)                    // 131328
#define SV_BYTES (128 * VLD * 2)                    // 18432
#define SMEM_BYTES (SS_BYTES + SV_BYTES + 16)       // + 4-entry reduce slab

#define ASYNC_CP16(off)                                                   \
  asm volatile("global_load_async_to_lds_b128 %0, %1, off offset:" #off   \
               :: "v"(ldsoff), "v"(gaddr) : "memory")

__global__ __launch_bounds__(128) void attn_kernel(
    const __bf16* __restrict__ Q, const __bf16* __restrict__ Km,
    const __bf16* __restrict__ V, const float* __restrict__ biasTab,
    __bf16* __restrict__ ctx, float* __restrict__ attn_out) {
  extern __shared__ char smem[];
  float*  sS   = (float*)smem;                       // [16][SSTR]
  __bf16* sV   = (__bf16*)(smem + SS_BYTES);         // [128][VLD]
  float*  sRed = (float*)(smem + SS_BYTES + SV_BYTES); // [4]

  const int bid = blockIdx.x;
  const int b   = bid >> 11;        // / (H * S/16)
  const int h   = (bid >> 7) & 15;
  const int qt  = bid & 127;
  const int q0  = qt * 16;
  const size_t tb = (size_t)b * S_;

  const int tid  = threadIdx.x;
  const int lane = tid & 31;
  const int wvid = tid >> 5;
  const int m    = lane & 15;
  const int half = lane >> 4;
  const int kb   = half * 8;
  const int ko   = half * 16;

  // ---- stage 1: scores = Q K^T + bias, into LDS -------------------------
  const __bf16* qrow = Q + (tb + q0 + m) * D_ + h * DK_;
  v16bf a0, a1;
#pragma unroll
  for (int j = 0; j < 8; ++j) {
    a0[j]     = qrow[kb + j];
    a0[8 + j] = qrow[16 + kb + j];
    a1[j]     = qrow[32 + kb + j];
    a1[8 + j] = qrow[48 + kb + j];
  }

  for (int kt = wvid; kt < S_ / 16; kt += 4) {
    // B fragment of K^T: two 16B pieces of a K row per lane (table k order).
    const __bf16* krow = Km + (tb + kt * 16 + m) * D_ + h * DK_ + kb;
    v8bfx k0 = *(const v8bfx*)(krow);
    v8bfx k1 = *(const v8bfx*)(krow + 16);
    v8bfx k2 = *(const v8bfx*)(krow + 32);
    v8bfx k3 = *(const v8bfx*)(krow + 48);
    v16bf b0, b1;
#pragma unroll
    for (int j = 0; j < 8; ++j) {
      b0[j] = k0[j]; b0[8 + j] = k1[j];
      b1[j] = k2[j]; b1[8 + j] = k3[j];
    }
    v8f c;
#pragma unroll
    for (int j = 0; j < 8; ++j) c[j] = 0.0f;
    c = wmma_bf16(a0, b0, c);
    c = wmma_bf16(a1, b1, c);
#pragma unroll
    for (int j = 0; j < 8; ++j) {
      int r = j + 8 * half;
      int kcol = kt * 16 + m;
      sS[r * SSTR + kcol] = c[j] + biasTab[h * 4096 + (kcol - (q0 + r) + 2047)];
    }
  }
  __syncthreads();

  // ---- stage 2: row softmax (shuffle + 4-slot LDS), write probs ---------
  for (int r = 0; r < 16; ++r) {
    float mx = -3.0e38f;
#pragma unroll
    for (int i = 0; i < 16; ++i) mx = fmaxf(mx, sS[r * SSTR + tid + 128 * i]);
    mx = wave_max(mx);
    if (lane == 0) sRed[wvid] = mx;
    __syncthreads();
    float rowmax = fmaxf(fmaxf(sRed[0], sRed[1]), fmaxf(sRed[2], sRed[3]));

    float lsum = 0.0f;
#pragma unroll
    for (int i = 0; i < 16; ++i) {
      int idx = r * SSTR + tid + 128 * i;
      float p = __expf(sS[idx] - rowmax);
      sS[idx] = p;
      lsum += p;
    }
    lsum = wave_sum(lsum);
    __syncthreads();                 // sRed max values consumed by everyone
    if (lane == 0) sRed[wvid] = lsum;
    __syncthreads();
    float inv = 1.0f / (sRed[0] + sRed[1] + sRed[2] + sRed[3]);

    size_t abase = (((size_t)(b * H_ + h)) * S_ + (q0 + r)) * S_;
#pragma unroll
    for (int i = 0; i < 16; ++i) {
      int kc = tid + 128 * i;        // coalesced across threads
      float pn = sS[r * SSTR + kc] * inv;
      sS[r * SSTR + kc] = pn;
      attn_out[abase + kc] = pn;
    }
    __syncthreads();                 // sRed sums consumed before next row
  }

  // ---- stage 3: ctx = P * V; V staged via async global->LDS DMA ---------
  const int nt = wvid;               // each wave owns 16 output dims
  v8f acc;
#pragma unroll
  for (int j = 0; j < 8; ++j) acc[j] = 0.0f;

  for (int ch = 0; ch < S_ / 128; ++ch) {
    __syncthreads();                 // sV free for reuse
    {  // each thread async-copies one 128B token row of V_h into LDS
      unsigned ldsoff = (unsigned)(uintptr_t)(&sV[tid * VLD]);
      unsigned long long gaddr =
          (unsigned long long)(uintptr_t)(V + (tb + ch * 128 + tid) * D_ + h * DK_);
      ASYNC_CP16(0);  ASYNC_CP16(16); ASYNC_CP16(32); ASYNC_CP16(48);
      ASYNC_CP16(64); ASYNC_CP16(80); ASYNC_CP16(96); ASYNC_CP16(112);
      asm volatile("s_wait_asynccnt 0x0" ::: "memory");
    }
    __syncthreads();
#pragma unroll
    for (int kt2 = 0; kt2 < 4; ++kt2) {
      int kl = kt2 * 32;
      v16bf af;
#pragma unroll
      for (int j = 0; j < 8; ++j) {
        af[j]     = (__bf16)sS[m * SSTR + ch * 128 + kl + kb + j];
        af[8 + j] = (__bf16)sS[m * SSTR + ch * 128 + kl + 16 + kb + j];
      }
      v16bf bf;
#pragma unroll
      for (int j = 0; j < 8; ++j) {
        bf[j]     = sV[(kl + kb + j) * VLD + nt * 16 + m];
        bf[8 + j] = sV[(kl + 16 + kb + j) * VLD + nt * 16 + m];
      }
      acc = wmma_bf16(af, bf, acc);
    }
  }

#pragma unroll
  for (int j = 0; j < 8; ++j) {
    int r = j + 8 * half;
    ctx[(tb + q0 + r) * D_ + h * DK_ + nt * 16 + m] = (__bf16)acc[j];
  }
}

// ---------------------------------------------------------------------------
// Workspace layout (bytes). Total ~48.3 MB.
#define OFF_X    ((size_t)0)                       // X bf16      8 MB
#define OFF_WQ   ((size_t)8  * 1024 * 1024)        // wq^T bf16   2 MB
#define OFF_WK   ((size_t)10 * 1024 * 1024)
#define OFF_WV   ((size_t)12 * 1024 * 1024)
#define OFF_WO   ((size_t)14 * 1024 * 1024)
#define OFF_Q    ((size_t)16 * 1024 * 1024)        // Q bf16      8 MB
#define OFF_K    ((size_t)24 * 1024 * 1024)
#define OFF_V    ((size_t)32 * 1024 * 1024)
#define OFF_CTX  ((size_t)40 * 1024 * 1024)
#define OFF_BIAS ((size_t)48 * 1024 * 1024)        // bias table 256 KB

extern "C" void kernel_launch(void* const* d_in, const int* in_sizes, int n_in,
                              void* d_out, int out_size, void* d_ws,
                              size_t ws_size, hipStream_t stream) {
  const float* hs  = (const float*)d_in[0];
  const float* wq  = (const float*)d_in[1];
  const float* wk  = (const float*)d_in[2];
  const float* wv  = (const float*)d_in[3];
  const float* wo  = (const float*)d_in[4];
  const float* rel = (const float*)d_in[5];

  float* out_attn_output = (float*)d_out;                       // [B,S,D]
  float* out_attn = out_attn_output + (size_t)B_ * S_ * D_;     // [B,H,S,S]

  char* ws = (char*)d_ws;
  __bf16* Xb  = (__bf16*)(ws + OFF_X);
  __bf16* Wqt = (__bf16*)(ws + OFF_WQ);
  __bf16* Wkt = (__bf16*)(ws + OFF_WK);
  __bf16* Wvt = (__bf16*)(ws + OFF_WV);
  __bf16* Wot = (__bf16*)(ws + OFF_WO);
  __bf16* Qb  = (__bf16*)(ws + OFF_Q);
  __bf16* Kb  = (__bf16*)(ws + OFF_K);
  __bf16* Vb  = (__bf16*)(ws + OFF_V);
  __bf16* Cxb = (__bf16*)(ws + OFF_CTX);
  float*  biasTab = (float*)(ws + OFF_BIAS);

  const int nX = B_ * S_ * D_;      // 4,194,304
  const int nW = D_ * H_ * DK_;     // 1,048,576
  cvt_bf16_kernel<<<(nX + 255) / 256, 256, 0, stream>>>(hs, Xb, nX);
  cvt_tr_bf16_kernel<<<(nW + 255) / 256, 256, 0, stream>>>(wq, Wqt, D_, H_ * DK_);
  cvt_tr_bf16_kernel<<<(nW + 255) / 256, 256, 0, stream>>>(wk, Wkt, D_, H_ * DK_);
  cvt_tr_bf16_kernel<<<(nW + 255) / 256, 256, 0, stream>>>(wv, Wvt, D_, H_ * DK_);
  cvt_tr_bf16_kernel<<<(nW + 255) / 256, 256, 0, stream>>>(wo, Wot, H_ * DK_, D_);

  bias_table_kernel<<<(H_ * 4095 + 255) / 256, 256, 0, stream>>>(rel, biasTab);

  dim3 gg((B_ * S_) / 128, (H_ * DK_) / 64);  // 32 x 16
  gemm_bf16_kernel<<<gg, 128, 0, stream>>>(Xb, Wqt, Qb, nullptr,
                                           B_ * S_, H_ * DK_, D_, 0);
  gemm_bf16_kernel<<<gg, 128, 0, stream>>>(Xb, Wkt, Kb, nullptr,
                                           B_ * S_, H_ * DK_, D_, 0);
  gemm_bf16_kernel<<<gg, 128, 0, stream>>>(Xb, Wvt, Vb, nullptr,
                                           B_ * S_, H_ * DK_, D_, 0);

  attn_kernel<<<B_ * H_ * (S_ / 16), 128, SMEM_BYTES, stream>>>(
      Qb, Kb, Vb, biasTab, Cxb, out_attn);

  dim3 go((B_ * S_) / 128, D_ / 64);
  gemm_bf16_kernel<<<go, 128, 0, stream>>>(Cxb, Wot, nullptr, out_attn_output,
                                           B_ * S_, D_, H_ * DK_, 1);
}
